// RetinaNet2_67937792688791
// MI455X (gfx1250) — matrix operand order repaired and checked by reading, not compiled, not run on hardware
//
#include <hip/hip_runtime.h>
#include <hip/hip_bf16.h>

typedef __attribute__((ext_vector_type(16))) _Float16     v16h;
typedef __attribute__((ext_vector_type(8)))  float        v8f;
typedef __attribute__((ext_vector_type(4)))  unsigned int v4u;
typedef __attribute__((ext_vector_type(4)))  int          v4i;
typedef __attribute__((ext_vector_type(8)))  int          v8i;

#define KDIM     2304          // 256 * 9
#define KT       72            // 2304 / 32
#define MT       6             // 96 padded rows: [0,24) reg, [32,86) cls
#define BCOLS    128           // positions per block (8 waves * 16)
#define AHWT     32760         // 6 * (4096+1024+256+64+16+4)
#define SB_STRIDE 48           // halves per im2col column (32 + pad, 96B = 32B aligned)
#define A_SLAB_DW 1536         // A slab per k-step: 6*32*16 halves = 6144B = 1536 dwords

#if defined(__has_builtin)
#if __has_builtin(__builtin_amdgcn_tensor_load_to_lds)
#define HAVE_TDM 1
#endif
#if __has_builtin(__builtin_amdgcn_sched_barrier)
#define HAVE_SCHEDBAR 1
#endif
#endif

__device__ __forceinline__ void sched_fence()
{
#ifdef HAVE_SCHEDBAR
    __builtin_amdgcn_sched_barrier(0);
#endif
}

// ---------------------------------------------------------------------------
// K ordering (tap-major): k' = tap*256 + c, tap = r*3+s.  Each 32-wide k-tile
// then has a single tap (256 % 32 == 0), making the im2col gather branch-light.
// Prep: pack fp32 conv weights into f16 WMMA A-fragment layout under k'.
// Layout: pk[((g2*KT + kt)*MT + mt)*512 + lane*16 + e]
//   m = mt*16 + (lane & 15), k' = kt*32 + (lane/16)*16 + e
//   source element: w[o*2304 + c*9 + tap],  c = k' & 255, tap = k' >> 8
// ---------------------------------------------------------------------------
__global__ void pack_weights_kernel(const float* __restrict__ wr1, const float* __restrict__ wc1,
                                    const float* __restrict__ wr2, const float* __restrict__ wc2,
                                    _Float16* __restrict__ pk)
{
    int t    = blockIdx.x * blockDim.x + threadIdx.x;
    int frag = t >> 5;
    int lane = t & 31;
    if (frag >= 2 * KT * MT) return;

    int g2  = frag / (KT * MT);
    int rem = frag % (KT * MT);
    int kt  = rem / MT;
    int mt  = rem % MT;
    const float* wr = g2 ? wr2 : wr1;
    const float* wc = g2 ? wc2 : wc1;

    int m = mt * 16 + (lane & 15);
    int g = lane >> 4;

    v16h out;
#pragma unroll
    for (int e = 0; e < 16; ++e) {
        int kp  = kt * 32 + g * 16 + e;   // permuted k
        int c   = kp & 255;
        int tap = kp >> 8;
        float v = 0.0f;
        if (mt < 2) {                      // reg head rows 0..31 (valid < 24)
            if (m < 24) v = wr[m * KDIM + c * 9 + tap];
        } else {                           // cls head rows 32..95 (valid < 86)
            int o = m - 32;
            if (o < 54) v = wc[o * KDIM + c * 9 + tap];
        }
        out[e] = (_Float16)v;
    }
    *(v16h*)(pk + (size_t)frag * 512 + lane * 16) = out;
}

#ifdef HAVE_TDM
// 1-D TDM copy: 1536 dwords global -> LDS (D# per CDNA5 ISA)
__device__ __forceinline__ void tdm_load_slab(const _Float16* __restrict__ src, unsigned lds_off)
{
    unsigned long long ga = (unsigned long long)(size_t)src;
    v4u g0 = { 1u,                                               // count=1 (valid user desc)
               lds_off,                                          // lds_addr [63:32]
               (unsigned)(ga & 0xFFFFFFFFu),                     // global_addr [95:64]
               (unsigned)(((ga >> 32) & 0x01FFFFFFu) | 0x80000000u) }; // addr[56:32] | type=2
    v8i g1 = { 0x00020000,                                       // data_size=4B; mask=0
               (int)(A_SLAB_DW << 16),                           // tensor_dim0 lo16 @ [31:16]
               0,                                                // tensor_dim0 hi, tensor_dim1 lo
               (int)(A_SLAB_DW << 16),                           // tile_dim0 @ [31:16]
               0,                                                // tile_dim1=0 (1D), tile_dim2=0
               A_SLAB_DW,                                        // tensor_dim0_stride lo32
               0, 0 };
    v4i gz = { 0, 0, 0, 0 };
#if __clang_major__ >= 23
    v8i gz8 = { 0, 0, 0, 0, 0, 0, 0, 0 };
    __builtin_amdgcn_tensor_load_to_lds(g0, g1, gz, gz, gz8, 0);
#else
    __builtin_amdgcn_tensor_load_to_lds(g0, g1, gz, gz, 0);
#endif
}
#endif

// ---------------------------------------------------------------------------
// Main: implicit-GEMM conv head for one pyramid level.
// Block = 8 waves: 96x128 output tile, K loop of 72 x (16x16x32) WMMA steps.
// ---------------------------------------------------------------------------
__global__ __launch_bounds__(256)
void retina_head_kernel(const float* __restrict__ x, const _Float16* __restrict__ pk,
                        const float* __restrict__ breg, const float* __restrict__ bcls,
                        float* __restrict__ out_loc, float* __restrict__ out_conf,
                        int H, int W, int lvl_off, int g2)
{
    __shared__ _Float16 sA[2][MT * 32 * 16];        // 2 x 6KB  weight slab (A frags)
    __shared__ _Float16 sB[2][BCOLS * SB_STRIDE];   // 2 x 12KB im2col slab [col][k]
    __shared__ float    sBias[96];

    const int tid   = threadIdx.x;
    const int HW    = H * W;
    const int Ptot  = 16 * HW;
    const int pbase = blockIdx.x * BCOLS;

    if (tid < 96) {
        float b = 0.0f;
        if (tid < 24)                   b = breg[tid];
        else if (tid >= 32 && tid < 86) b = bcls[tid - 32];
        sBias[tid] = b;
    }

    // ---- staging-role indices: thread t owns column (t&127), K-half (t>>7) ----
    const int  scol   = tid & 127;
    const int  skh    = tid >> 7;              // 0..1
    const int  pcol   = pbase + scol;
    const bool cvalid = pcol < Ptot;
    int n = 0, h = 0, w = 0;
    if (cvalid) { n = pcol / HW; int r = pcol - n * HW; h = r / W; w = r - h * W; }
    const float* xn = x + (size_t)n * 256 * HW;

    const _Float16* apk = pk + (size_t)g2 * KT * MT * 512;

    // ---- compute-role indices ----
    const int wave = tid >> 5;
    const int lane = tid & 31;
    const int ln   = lane & 15;
    const int g    = lane >> 4;
    const int bcol = wave * 16 + ln;           // this wave's B-fragment column

    v8f acc[MT];
#pragma unroll
    for (int i = 0; i < MT; ++i) acc[i] = (v8f){0, 0, 0, 0, 0, 0, 0, 0};

    // im2col staging under tap-major k'.  All 16 gathers use a CLAMPED (always
    // valid) address so they issue as one unconditional clause; the halo/pad
    // zero-mask is applied afterwards with cndmask (no exec branching).
    auto stageB = [&](int kt, int buf) {
        int tap = kt >> 3;                     // 0..8, uniform
        int dr  = tap / 3, dc = tap - dr * 3;
        int hh  = h + dr - 1, ww = w + dc - 1;
        bool ok = cvalid & ((unsigned)hh < (unsigned)H) & ((unsigned)ww < (unsigned)W);
        int hc  = min(max(hh, 0), H - 1);
        int wc  = min(max(ww, 0), W - 1);
        const float* base = xn + (ptrdiff_t)((kt & 7) * 32 + skh * 16) * HW + (hc * W + wc);
        float f[16];
#pragma unroll
        for (int j = 0; j < 16; ++j)           // independent -> one s_clause
            f[j] = base[(ptrdiff_t)j * HW];
        _Float16* dst = &sB[buf][scol * SB_STRIDE + skh * 16];
#pragma unroll
        for (int jj = 0; jj < 16; jj += 2) {
            float f0 = ok ? f[jj]     : 0.0f;
            float f1 = ok ? f[jj + 1] : 0.0f;
            _Float16 h0 = (_Float16)f0, h1 = (_Float16)f1;
            unsigned int bits = (unsigned int)__builtin_bit_cast(unsigned short, h0) |
                                ((unsigned int)__builtin_bit_cast(unsigned short, h1) << 16);
            *(unsigned int*)&dst[jj] = bits;
        }
    };

#ifndef HAVE_TDM
    auto stageA = [&](int kt, int buf) {       // fallback: cooperative dword copy
        const unsigned int* src = (const unsigned int*)(apk + (size_t)kt * (MT * 512));
        unsigned int*       dst = (unsigned int*)&sA[buf][0];
#pragma unroll
        for (int i = 0; i < 6; ++i)
            dst[tid + i * 256] = src[tid + i * 256];
    };
#endif

#ifdef HAVE_TDM
    const unsigned ldsA0 = (unsigned)(size_t)&sA[0][0];
    const unsigned ldsA1 = (unsigned)(size_t)&sA[1][0];
    if (wave == 0)                              // one DMA per block (TDM ignores EXEC)
        tdm_load_slab(apk, ldsA0);
#else
    stageA(0, 0);
#endif
    stageB(0, 0);
#ifdef HAVE_TDM
    if (wave == 0) __builtin_amdgcn_s_wait_tensorcnt((short)0);
#endif
    __syncthreads();

    for (int kt = 0; kt < KT; ++kt) {
        const int cur = kt & 1;

        // 1. start next A-slab DMA as early as possible
#ifdef HAVE_TDM
        if (wave == 0 && kt + 1 < KT)
            tdm_load_slab(apk + (size_t)(kt + 1) * (MT * 512), cur ? ldsA0 : ldsA1);
#endif

        // 2. fragment DS loads for this step (distinct registers)
        v16h bf = *(const v16h*)&sB[cur][bcol * SB_STRIDE + g * 16];
        v16h af[MT];
#pragma unroll
        for (int mt = 0; mt < MT; ++mt)
            af[mt] = *(const v16h*)&sA[cur][mt * 512 + lane * 16];

        // 3. im2col gather for next step (global clause overlaps DS latency)
        if (kt + 1 < KT) {
#ifndef HAVE_TDM
            stageA(kt + 1, cur ^ 1);
#endif
            stageB(kt + 1, cur ^ 1);
        }

        // 4. matrix ops back-to-back; nothing may be interleaved below this point
        sched_fence();
#pragma unroll
        for (int mt = 0; mt < MT; ++mt)
            acc[mt] = __builtin_amdgcn_wmma_f32_16x16x32_f16(
                false, af[mt], false, bf, (short)0, acc[mt], false, false);

#ifdef HAVE_TDM
        if (wave == 0 && kt + 1 < KT) __builtin_amdgcn_s_wait_tensorcnt((short)0);
#endif
        __syncthreads();
    }

    // ---- epilogue: bias + scatter per reference reshape/concat layout ----
    const int pe = pbase + wave * 16 + ln;
    if (pe < Ptot) {
        int n2 = pe / HW;
        int r2 = pe - n2 * HW;
        int h2 = r2 / W;
        int w2 = r2 - h2 * W;
        int sp = lvl_off + h2 * W + w2;    // + a*HW added per element
#pragma unroll
        for (int mt = 0; mt < MT; ++mt) {
#pragma unroll
            for (int v = 0; v < 8; ++v) {
                int   m   = mt * 16 + g * 8 + v;
                float val = acc[mt][v] + sBias[m];
                if (m < 24) {                              // bbox_delta [N,4,AHWT]
                    int c4 = m / 6, a = m - c4 * 6;
                    out_loc[(size_t)n2 * (4 * AHWT) + c4 * AHWT + sp + a * HW] = val;
                } else if (m >= 32 && m < 86) {            // confs [N,9,AHWT]
                    int o = m - 32;
                    int kc = o / 6, a = o - kc * 6;
                    out_conf[(size_t)n2 * (9 * AHWT) + kc * AHWT + sp + a * HW] = val;
                }
            }
        }
    }
}

// ---------------------------------------------------------------------------
extern "C" void kernel_launch(void* const* d_in, const int* in_sizes, int n_in,
                              void* d_out, int out_size, void* d_ws, size_t ws_size,
                              hipStream_t stream)
{
    const float* feats[6];
    for (int i = 0; i < 6; ++i) feats[i] = (const float*)d_in[i];
    const float* wr1 = (const float*)d_in[6];
    const float* br1 = (const float*)d_in[7];
    const float* wc1 = (const float*)d_in[8];
    const float* bc1 = (const float*)d_in[9];
    const float* wr2 = (const float*)d_in[10];
    const float* br2 = (const float*)d_in[11];
    const float* wc2 = (const float*)d_in[12];
    const float* bc2 = (const float*)d_in[13];

    _Float16* pk       = (_Float16*)d_ws;              // 884,736 B of packed weights
    float*    out_loc  = (float*)d_out;
    float*    out_conf = out_loc + (size_t)16 * 4 * AHWT;

    // 864 fragments * 32 lanes = 27648 threads
    pack_weights_kernel<<<108, 256, 0, stream>>>(wr1, wc1, wr2, wc2, pk);

    static const int Hs[6]   = {64, 32, 16, 8, 4, 2};
    static const int offs[6] = {0, 24576, 30720, 32256, 32640, 32736};
    for (int l = 0; l < 6; ++l) {
        int H = Hs[l], W = Hs[l];
        int Ptot   = 16 * H * W;
        int blocks = (Ptot + BCOLS - 1) / BCOLS;
        int g2 = (l >= 2) ? 1 : 0;
        const float* br = g2 ? br2 : br1;
        const float* bc = g2 ? bc2 : bc1;
        retina_head_kernel<<<blocks, 256, 0, stream>>>(
            feats[l], pk, br, bc, out_loc, out_conf, H, W, offs[l], g2);
    }
}
